// spatialAttention_54468775247910
// MI455X (gfx1250) — matrix operand order repaired and checked by reading, not compiled, not run on hardware
//
#include <hip/hip_runtime.h>
#include <hip/hip_bf16.h>

typedef __attribute__((ext_vector_type(16))) _Float16 v16h;
typedef __attribute__((ext_vector_type(8)))  _Float16 v8h;
typedef __attribute__((ext_vector_type(8)))  float    v8f;

#define MODE_F16     0   // D + bias[c]            -> f16 [M,N]
#define MODE_KT      1   // D + bias[c]            -> f16 kT layout [bt,kh,dd,N]
#define MODE_F32_RES 2   // D + bias[c] + resid    -> f32 [M,N]
#define MODE_RELU16  3   // relu(D + bias[c])      -> f16 [M,N]

static __device__ inline v16h frag_combine(v8h lo, v8h hi) {
    v16h r;
#pragma unroll
    for (int i = 0; i < 8; ++i) { r[i] = lo[i]; r[i + 8] = hi[i]; }
    return r;
}

// A-matrix 16x32 f16 fragment from row-major [*, lda] memory (or LDS).
// lane l: row = l&15 ; halves 0-7 = K[kb..kb+7], halves 8-15 = K[kb+16..kb+23], kb = (l>=16)?8:0
static __device__ inline v16h load_frag_a(const _Float16* __restrict__ base, int lda, int k0) {
    int lane = threadIdx.x & 31;
    int m    = lane & 15;
    int kb   = (lane >> 4) << 3;
    const _Float16* p = base + (size_t)m * lda + k0 + kb;
    v8h lo = *(const v8h*)p;
    v8h hi = *(const v8h*)(p + 16);
    return frag_combine(lo, hi);
}

// B-matrix 32x16 f16 fragment from row-major [K, ldb] memory (or LDS panel).
// lane l: col = n0 + (l&15) ; halves 0-7 = K[kb..kb+7], halves 8-15 = K[kb+16..kb+23], kb = k0 + ((l>=16)?8:0)
static __device__ inline v16h load_frag_b(const _Float16* __restrict__ base, int ldb, int k0, int n0) {
    int lane = threadIdx.x & 31;
    int kb   = (lane >> 4) << 3;
    const _Float16* p = base + (size_t)(k0 + kb) * ldb + n0 + (lane & 15);
    v16h r;
#pragma unroll
    for (int i = 0; i < 8; ++i) { r[i] = p[(size_t)i * ldb]; r[i + 8] = p[(size_t)(i + 16) * ldb]; }
    return r;
}

static __device__ inline v8f wmma_f16f32(v16h a, v16h b, v8f c) {
    return __builtin_amdgcn_wmma_f32_16x16x32_f16(false, a, false, b, (short)0, c, false, false);
}

// ---------------------------------------------------------------------------
// Weight transpose+convert: src [R,C] f32 -> dst [C,R] f16 (dst[c*R+r]=src[r*C+c])
// ---------------------------------------------------------------------------
__global__ void transpose_f16_kernel(const float* __restrict__ src, _Float16* __restrict__ dst,
                                     int R, int C) {
    int t = blockIdx.x * blockDim.x + threadIdx.x;
    if (t >= R * C) return;
    int r = t / C, c = t % C;
    dst[(size_t)c * R + r] = (_Float16)src[(size_t)r * C + c];
}

// ---------------------------------------------------------------------------
// Graph bias: bias[i][c] = alpha*softmax_row(relu(E E^T)) + beta*lap + (-1e9 if lap==0)
// ---------------------------------------------------------------------------
__global__ void graph_bias_kernel(const float* __restrict__ lap, const float* __restrict__ emb,
                                  const float* __restrict__ alphaP, const float* __restrict__ betaP,
                                  float* __restrict__ bias) {
    __shared__ float sE[64];
    __shared__ float red[256];
    int i = blockIdx.x, t = threadIdx.x;
    if (t < 64) sE[t] = emb[(size_t)i * 64 + t];
    __syncthreads();
    float s0, s1;
    {
        const float* er = emb + (size_t)t * 64;
        float d = 0.f;
#pragma unroll 8
        for (int e = 0; e < 64; ++e) d += sE[e] * er[e];
        s0 = d > 0.f ? d : 0.f;
        er = emb + (size_t)(t + 256) * 64;
        d = 0.f;
#pragma unroll 8
        for (int e = 0; e < 64; ++e) d += sE[e] * er[e];
        s1 = d > 0.f ? d : 0.f;
    }
    red[t] = fmaxf(s0, s1); __syncthreads();
    for (int o = 128; o; o >>= 1) { if (t < o) red[t] = fmaxf(red[t], red[t + o]); __syncthreads(); }
    float mx = red[0]; __syncthreads();
    float e0 = __expf(s0 - mx), e1 = __expf(s1 - mx);
    red[t] = e0 + e1; __syncthreads();
    for (int o = 128; o; o >>= 1) { if (t < o) red[t] += red[t + o]; __syncthreads(); }
    float rs = 1.0f / red[0];
    float alpha = alphaP[0], beta = betaP[0];
    {
        float lv = lap[(size_t)i * 512 + t];
        bias[(size_t)i * 512 + t] = alpha * e0 * rs + beta * lv + ((lv != 0.f) ? 0.f : -1e9f);
        lv = lap[(size_t)i * 512 + t + 256];
        bias[(size_t)i * 512 + t + 256] = alpha * e1 * rs + beta * lv + ((lv != 0.f) ? 0.f : -1e9f);
    }
}

// ---------------------------------------------------------------------------
// LayerNorm over last dim (512), one block per row, f32 in -> f16 out
// ---------------------------------------------------------------------------
__global__ void layernorm_kernel(const float* __restrict__ x, const float* __restrict__ g,
                                 const float* __restrict__ b, _Float16* __restrict__ out) {
    __shared__ float red[256];
    int row = blockIdx.x, t = threadIdx.x;
    const float* xr = x + (size_t)row * 512;
    float v0 = xr[t], v1 = xr[t + 256];
    red[t] = v0 + v1; __syncthreads();
    for (int o = 128; o; o >>= 1) { if (t < o) red[t] += red[t + o]; __syncthreads(); }
    float mean = red[0] * (1.f / 512.f); __syncthreads();
    red[t] = v0 * v0 + v1 * v1; __syncthreads();
    for (int o = 128; o; o >>= 1) { if (t < o) red[t] += red[t + o]; __syncthreads(); }
    float var  = red[0] * (1.f / 512.f) - mean * mean;
    float rstd = rsqrtf(var + 1e-5f);
    _Float16* orow = out + (size_t)row * 512;
    orow[t]       = (_Float16)((v0 - mean) * rstd * g[t] + b[t]);
    orow[t + 256] = (_Float16)((v1 - mean) * rstd * g[t + 256] + b[t + 256]);
}

// ---------------------------------------------------------------------------
// LDS-staged WMMA GEMM: C[M,N] = A[M,K] @ B[K,N] (+ epilogue).
// Block = 256 threads (8 waves) -> 128x64 tile; shared 64-col B panel in LDS,
// double-buffered 64-deep K chunks. All four B fragments are materialized into
// distinct locals before the WMMAs so ds_loads stay in flight (no dscnt-0 drains).
// ---------------------------------------------------------------------------
#define BPAD 72   // 64 + 8 halves row padding (16 B) to spread LDS banks

__global__ void gemm_wmma_kernel(const _Float16* __restrict__ A, int lda,
                                 const _Float16* __restrict__ Bm, int ldb,
                                 int N, int Ksz, int mode,
                                 const float* __restrict__ bias,
                                 const float* __restrict__ resid,
                                 float* __restrict__ outF,
                                 _Float16* __restrict__ outH) {
    __shared__ __attribute__((aligned(16))) _Float16 sB[2][64][BPAD];

    int t  = threadIdx.x;
    int w  = t >> 5;
    int c0 = blockIdx.x << 6;                       // 64-wide N strip
    int r0 = (blockIdx.y << 7) + (w << 4);          // this wave's 16 rows
    const _Float16* Abase = A + (size_t)r0 * lda;

    // stage first K chunk (kc = 0): 64 rows x 64 halves = 512 x 16B chunks
#pragma unroll
    for (int i = 0; i < 2; ++i) {
        int c = t + (i << 8);
        int row = c >> 3, col8 = (c & 7) << 3;
        *(v8h*)&sB[0][row][col8] = *(const v8h*)(Bm + (size_t)row * ldb + c0 + col8);
    }
    __syncthreads();

    v8f acc0 = {}, acc1 = {}, acc2 = {}, acc3 = {};
    int parity = 0;
    for (int kc = 0; kc < Ksz; kc += 64) {
        if (kc + 64 < Ksz) {                        // prefetch next chunk into other buffer
            int nb = parity ^ 1;
#pragma unroll
            for (int i = 0; i < 2; ++i) {
                int c = t + (i << 8);
                int row = c >> 3, col8 = (c & 7) << 3;
                *(v8h*)&sB[nb][row][col8] =
                    *(const v8h*)(Bm + (size_t)(kc + 64 + row) * ldb + c0 + col8);
            }
        }
        const _Float16* bp = &sB[parity][0][0];

        // both A fragments + all B fragments in distinct locals -> overlapped loads
        v16h a0 = load_frag_a(Abase, lda, kc);
        v16h a1 = load_frag_a(Abase, lda, kc + 32);

        v16h b00 = load_frag_b(bp, BPAD, 0, 0);
        v16h b01 = load_frag_b(bp, BPAD, 0, 16);
        v16h b02 = load_frag_b(bp, BPAD, 0, 32);
        v16h b03 = load_frag_b(bp, BPAD, 0, 48);
        acc0 = wmma_f16f32(a0, b00, acc0);
        acc1 = wmma_f16f32(a0, b01, acc1);
        acc2 = wmma_f16f32(a0, b02, acc2);
        acc3 = wmma_f16f32(a0, b03, acc3);

        v16h b10 = load_frag_b(bp, BPAD, 32, 0);
        v16h b11 = load_frag_b(bp, BPAD, 32, 16);
        v16h b12 = load_frag_b(bp, BPAD, 32, 32);
        v16h b13 = load_frag_b(bp, BPAD, 32, 48);
        acc0 = wmma_f16f32(a1, b10, acc0);
        acc1 = wmma_f16f32(a1, b11, acc1);
        acc2 = wmma_f16f32(a1, b12, acc2);
        acc3 = wmma_f16f32(a1, b13, acc3);

        __syncthreads();
        parity ^= 1;
    }

    int lane = t & 31;
    int n    = lane & 15;
    int mb   = (lane >> 4) << 3;
    v8f accs[4] = {acc0, acc1, acc2, acc3};
#pragma unroll
    for (int j = 0; j < 4; ++j) {
        int   c  = c0 + (j << 4) + n;
        float bv = bias ? bias[c] : 0.f;
#pragma unroll
        for (int i = 0; i < 8; ++i) {
            int   r   = r0 + mb + i;
            float val = accs[j][i] + bv;
            if (mode == MODE_F16) {
                outH[(size_t)r * N + c] = (_Float16)val;
            } else if (mode == MODE_KT) {                // [bt, kh, dd, node]
                int bt = r >> 9, node = r & 511;
                int kh = c >> 6, dd = c & 63;
                outH[(((size_t)(bt * 8 + kh) * 64 + dd) << 9) + node] = (_Float16)val;
            } else if (mode == MODE_F32_RES) {
                outF[(size_t)r * N + c] = val + resid[(size_t)r * N + c];
            } else {                                     // MODE_RELU16
                outH[(size_t)r * N + c] = (_Float16)(val > 0.f ? val : 0.f);
            }
        }
    }
}

// ---------------------------------------------------------------------------
// Fused attention per (bt, head, 16-row block): WMMA logits -> LDS softmax ->
// WMMA P@V from LDS, scaled by 1/rowsum. 128 threads (4 waves).
// ---------------------------------------------------------------------------
__global__ void attention_kernel(const _Float16* __restrict__ q,
                                 const _Float16* __restrict__ kT,   // [bt,kh,dd,N]
                                 const _Float16* __restrict__ v,    // [bt*N, 512]
                                 const float*    __restrict__ bias, // [N,N]
                                 _Float16*       __restrict__ outH) // [bt*N, 512]
{
    __shared__ float                                  s_logits[16][512];
    __shared__ __attribute__((aligned(16))) _Float16 s_probs[16][512];
    __shared__ float s_red[16][8];
    __shared__ float s_rowmax[16];
    __shared__ float s_rowsum[16];

    int rowblk = blockIdx.x;          // 0..31
    int kh     = blockIdx.y;          // 0..7
    int bt     = blockIdx.z;          // 0..23
    int t      = threadIdx.x;
    int w      = t >> 5;              // wave 0..3
    int lane   = t & 31;
    int jn     = lane & 15;
    int mb     = (lane >> 4) << 3;
    int rowbase = rowblk << 4;

    const _Float16* qbase  = q  + ((size_t)(bt * 512 + rowbase)) * 512 + kh * 64;
    const _Float16* ktHead = kT + ((size_t)(bt * 8 + kh)) * 64 * 512;

    // Q strip fragments (d=64 -> two K-steps), kept in registers all phase 1
    v16h a0 = load_frag_a(qbase, 512, 0);
    v16h a1 = load_frag_a(qbase, 512, 32);

    // Phase 1: logits strip 16x512 (wave w covers columns [w*128, w*128+128)),
    // two tiles per iteration with distinct B locals to keep loads in flight
    for (int tt = 0; tt < 4; ++tt) {
        int m0 = (w << 7) + (tt << 5);
        v16h b00 = load_frag_b(ktHead, 512, 0,  m0);
        v16h b01 = load_frag_b(ktHead, 512, 32, m0);
        v16h b10 = load_frag_b(ktHead, 512, 0,  m0 + 16);
        v16h b11 = load_frag_b(ktHead, 512, 32, m0 + 16);
        v8f accA = {}, accB = {};
        accA = wmma_f16f32(a0, b00, accA);
        accA = wmma_f16f32(a1, b01, accA);
        accB = wmma_f16f32(a0, b10, accB);
        accB = wmma_f16f32(a1, b11, accB);
#pragma unroll
        for (int i = 0; i < 8; ++i) {
            int rr = mb + i;
            int cc = m0 + jn;
            s_logits[rr][cc]      = accA[i] * 0.125f + bias[(size_t)(rowbase + rr) * 512 + cc];
            s_logits[rr][cc + 16] = accB[i] * 0.125f + bias[(size_t)(rowbase + rr) * 512 + cc + 16];
        }
    }
    __syncthreads();

    // Phase 2: softmax (store unnormalized exp as f16; fold 1/sum into epilogue)
    {
        int row = t >> 3, seg = t & 7, cb = seg << 6;
        float lmax = -3.4e38f;
        for (int c = 0; c < 64; ++c) lmax = fmaxf(lmax, s_logits[row][cb + c]);
        s_red[row][seg] = lmax; __syncthreads();
        if (seg == 0) {
            float m2 = s_red[row][0];
#pragma unroll
            for (int u = 1; u < 8; ++u) m2 = fmaxf(m2, s_red[row][u]);
            s_rowmax[row] = m2;
        }
        __syncthreads();
        float rmx = s_rowmax[row], lsum = 0.f;
        for (int c = 0; c < 64; ++c) {
            float e = __expf(s_logits[row][cb + c] - rmx);
            s_probs[row][cb + c] = (_Float16)e;
            lsum += e;
        }
        s_red[row][seg] = lsum; __syncthreads();
        if (seg == 0) {
            float su = 0.f;
#pragma unroll
            for (int u = 0; u < 8; ++u) su += s_red[row][u];
            s_rowsum[row] = su;
        }
        __syncthreads();
    }

    // Phase 3: out(16 x 64) = P(16x512) @ V(512x64); wave w covers dd [w*16, w*16+16)
    // two k-steps per iteration with distinct locals (LDS-A + global-B overlap)
    {
        int dd0 = w << 4;
        const _Float16* vHead = v + (size_t)bt * 512 * 512 + kh * 64;
        v8f acc = {};
        for (int k0 = 0; k0 < 512; k0 += 64) {
            v16h pa0 = load_frag_a(&s_probs[0][0], 512, k0);        // ds_load_b128
            v16h vb0 = load_frag_b(vHead, 512, k0, dd0);
            v16h pa1 = load_frag_a(&s_probs[0][0], 512, k0 + 32);
            v16h vb1 = load_frag_b(vHead, 512, k0 + 32, dd0);
            acc = wmma_f16f32(pa0, vb0, acc);
            acc = wmma_f16f32(pa1, vb1, acc);
        }
#pragma unroll
        for (int i = 0; i < 8; ++i) {
            int rr = mb + i;
            float val = acc[i] / s_rowsum[rr];
            outH[((size_t)(bt * 512 + rowbase + rr)) * 512 + kh * 64 + dd0 + jn] = (_Float16)val;
        }
    }
}

// ---------------------------------------------------------------------------
extern "C" void kernel_launch(void* const* d_in, const int* in_sizes, int n_in,
                              void* d_out, int out_size, void* d_ws, size_t ws_size,
                              hipStream_t stream) {
    const float* x    = (const float*)d_in[0];
    const float* lap  = (const float*)d_in[1];
    const float* emb  = (const float*)d_in[2];
    const float* Wq   = (const float*)d_in[3];  const float* bq  = (const float*)d_in[4];
    const float* Wk   = (const float*)d_in[5];  const float* bk  = (const float*)d_in[6];
    const float* Wv   = (const float*)d_in[7];  const float* bv  = (const float*)d_in[8];
    const float* Wo   = (const float*)d_in[9];  const float* bo  = (const float*)d_in[10];
    const float* W1   = (const float*)d_in[11]; const float* b1  = (const float*)d_in[12];
    const float* W2   = (const float*)d_in[13]; const float* b2  = (const float*)d_in[14];
    const float* g1   = (const float*)d_in[15]; const float* be1 = (const float*)d_in[16];
    const float* g2   = (const float*)d_in[17]; const float* be2 = (const float*)d_in[18];
    const float* alphaP = (const float*)d_in[19];
    const float* betaP  = (const float*)d_in[20];
    (void)in_sizes; (void)n_in; (void)out_size; (void)ws_size;

    const int Mrows = 24 * 512;   // B*T*N = 12288
    const int Dm = 512, D4 = 2048;

    char* wsb = (char*)d_ws;
    size_t off = 0;
    auto alloc = [&](size_t bytes) -> void* {
        off = (off + 255) & ~(size_t)255;
        void* p = wsb + off;
        off += bytes;
        return p;
    };

    _Float16* wqT  = (_Float16*)alloc((size_t)Dm * Dm * 2);
    _Float16* wkT  = (_Float16*)alloc((size_t)Dm * Dm * 2);
    _Float16* wvT  = (_Float16*)alloc((size_t)Dm * Dm * 2);
    _Float16* woT  = (_Float16*)alloc((size_t)Dm * Dm * 2);
    _Float16* w1T  = (_Float16*)alloc((size_t)Dm * D4 * 2);
    _Float16* w2T  = (_Float16*)alloc((size_t)D4 * Dm * 2);
    _Float16* xn   = (_Float16*)alloc((size_t)Mrows * Dm * 2);   // reused as attnout
    _Float16* qf   = (_Float16*)alloc((size_t)Mrows * Dm * 2);   // reused as h
    _Float16* ktf  = (_Float16*)alloc((size_t)Mrows * Dm * 2);
    _Float16* vf   = (_Float16*)alloc((size_t)Mrows * Dm * 2);
    float*    biasNN = (float*)alloc((size_t)512 * 512 * 4);
    float*    x1   = (float*)alloc((size_t)Mrows * Dm * 4);
    _Float16* ffn1 = (_Float16*)alloc((size_t)Mrows * D4 * 2);
    _Float16* attnout = xn;   // xn dead after QKV GEMMs
    _Float16* hbuf    = qf;   // q dead after attention

    // 1) weight transpose+convert to f16
    auto tgrid = [](int n) { return (n + 255) / 256; };
    transpose_f16_kernel<<<tgrid(Dm * Dm), 256, 0, stream>>>(Wq, wqT, Dm, Dm);
    transpose_f16_kernel<<<tgrid(Dm * Dm), 256, 0, stream>>>(Wk, wkT, Dm, Dm);
    transpose_f16_kernel<<<tgrid(Dm * Dm), 256, 0, stream>>>(Wv, wvT, Dm, Dm);
    transpose_f16_kernel<<<tgrid(Dm * Dm), 256, 0, stream>>>(Wo, woT, Dm, Dm);
    transpose_f16_kernel<<<tgrid(D4 * Dm), 256, 0, stream>>>(W1, w1T, D4, Dm);  // -> [512,2048]
    transpose_f16_kernel<<<tgrid(Dm * D4), 256, 0, stream>>>(W2, w2T, Dm, D4);  // -> [2048,512]

    // 2) graph bias
    graph_bias_kernel<<<512, 256, 0, stream>>>(lap, emb, alphaP, betaP, biasNN);

    // 3) LN1
    layernorm_kernel<<<Mrows, 256, 0, stream>>>(x, g1, be1, xn);

    // 4) QKV GEMMs (block = 128x64 tile, LDS-staged B)
    auto ggrid = [](int M, int N) { return dim3((unsigned)(N >> 6), (unsigned)(M >> 7)); };
    gemm_wmma_kernel<<<ggrid(Mrows, Dm), 256, 0, stream>>>(xn, Dm, wqT, Dm, Dm, Dm,
                                                           MODE_F16, bq, nullptr, nullptr, qf);
    gemm_wmma_kernel<<<ggrid(Mrows, Dm), 256, 0, stream>>>(xn, Dm, wkT, Dm, Dm, Dm,
                                                           MODE_KT, bk, nullptr, nullptr, ktf);
    gemm_wmma_kernel<<<ggrid(Mrows, Dm), 256, 0, stream>>>(xn, Dm, wvT, Dm, Dm, Dm,
                                                           MODE_F16, bv, nullptr, nullptr, vf);

    // 5) fused attention
    dim3 agrid(32, 8, 24);
    attention_kernel<<<agrid, 128, 0, stream>>>(qf, ktf, vf, biasNN, attnout);

    // 6) Wo GEMM + residual -> x1 (fp32)
    gemm_wmma_kernel<<<ggrid(Mrows, Dm), 256, 0, stream>>>(attnout, Dm, woT, Dm, Dm, Dm,
                                                           MODE_F32_RES, bo, x, x1, nullptr);

    // 7) LN2 -> h
    layernorm_kernel<<<Mrows, 256, 0, stream>>>(x1, g2, be2, hbuf);

    // 8) FFN1: relu(h @ W1^T + b1)
    gemm_wmma_kernel<<<ggrid(Mrows, D4), 256, 0, stream>>>(hbuf, Dm, w1T, D4, D4, Dm,
                                                           MODE_RELU16, b1, nullptr, nullptr, ffn1);

    // 9) FFN2: x1 + ffn1 @ W2^T + b2 -> d_out (fp32)
    gemm_wmma_kernel<<<ggrid(Mrows, Dm), 256, 0, stream>>>(ffn1, D4, w2T, Dm, Dm, D4,
                                                           MODE_F32_RES, b2, x1, (float*)d_out, nullptr);
}
